// ViT_76252849373615
// MI455X (gfx1250) — compile-verified, matrix operand order checked by reading, not run on hardware
//
#include <hip/hip_runtime.h>
#include <math.h>

// ---------------------------------------------------------------------------
// ViT-Base forward for MI455X (gfx1250, wave32).
// Big GEMMs (~2.16 TFLOP) -> v_wmma_f32_16x16x32_bf16, f32 accumulate,
// double-buffered LDS tiles filled with GLOBAL_LOAD_ASYNC_TO_LDS_B128
// (ASYNCcnt) when the toolchain exposes the builtin.
// ---------------------------------------------------------------------------

typedef __bf16 bf16;
typedef __attribute__((ext_vector_type(16))) __bf16 v16bf;
typedef __attribute__((ext_vector_type(8)))  __bf16 v8bf;
typedef __attribute__((ext_vector_type(8)))  float  v8f;

#if defined(__gfx1250__) && __has_builtin(__builtin_amdgcn_global_load_async_to_lds_b128)
#define USE_ASYNC 1
#define AS1 __attribute__((address_space(1)))
#define AS3 __attribute__((address_space(3)))
// builtin arg0 type per hipcc diagnostic: int __vector_size(16) in AS1
typedef int gv4i __attribute__((vector_size(4 * sizeof(int))));
#else
#define USE_ASYNC 0
#endif

__device__ __forceinline__ void wait_async0() {
#if USE_ASYNC
#if __has_builtin(__builtin_amdgcn_s_wait_asynccnt)
  __builtin_amdgcn_s_wait_asynccnt(0);
#else
  asm volatile("s_wait_asynccnt 0" ::: "memory");
#endif
#endif
}

constexpr int BATCH = 64, CH = 3, IMG = 224, PATCH = 16;
constexpr int DIM = 768, DEPTH = 12, HEADS = 12, HID = 3072, NCLS = 1000;
constexpr int HP = IMG / PATCH;          // 14
constexpr int NPATCH = HP * HP;          // 196
constexpr int PD = CH * PATCH * PATCH;   // 768
constexpr int HD = DIM / HEADS;          // 64
constexpr int NTOK = NPATCH + 1;         // 197
constexpr int MTOK = BATCH * NTOK;       // 12608 (mult of 64)
constexpr int MPATCH = BATCH * NPATCH;   // 12544 (mult of 64)

// ---------------------------------------------------------------------------
// WMMA GEMM: C[MxN](f32) = A[MxK](bf16 row-major) * Bt[NxK](bf16 row-major,
// i.e. the K x N weight pre-transposed by the convert kernel).
// 256 threads = 8 waves (2x4 wave grid); block tile 64x256; wave tile 32x64
// = 2x4 fragments of 16x16; K-step 32. LDS row stride 40 bf16 (80 B = 5*16B)
// -> 16B-aligned, bank-conflict-free ds_load_b128 fragment reads.
// Double-buffered tiles, async global->LDS, one barrier per K-step.
// MODE: 0 C=acc ; 1 C=acc+bias ; 2 C=gelu(acc+bias) ; 3 C+=acc+bias
// ---------------------------------------------------------------------------
constexpr int LDT = 40;
constexpr int BM = 64, BN = 256;

template <int MODE, bool OUTBF>
__global__ __launch_bounds__(256) void gemm_bf16_wmma(
    const bf16* __restrict__ A, const bf16* __restrict__ Bt,
    const float* __restrict__ bias, float* __restrict__ Cf,
    bf16* __restrict__ Cb, int M, int N, int K) {
  __shared__ __align__(16) bf16 As[2][BM * LDT];
  __shared__ __align__(16) bf16 Bs[2][BN * LDT];

  const int tid = threadIdx.x, lane = tid & 31, wave = tid >> 5;
  const int wr = wave >> 2, wc = wave & 3;  // 2 x 4 wave grid
  const int bm = blockIdx.y * BM, bn = blockIdx.x * BN;

  v8f acc[2][4] = {};

  // contiguous 16B-chunk copy coords (8 bf16 per chunk)
  const int crow = tid >> 2;        // 0..63
  const int ckof = (tid & 3) * 8;   // 0,8,16,24 elements

  // fragment coords per ISA 7.12.2 (16-bit 16x32 operand layout)
  const int lk = (lane < 16) ? 0 : 8;
  const int lm = lane & 15;

  auto issue_copy = [&](int buf, int kb) {
#if USE_ASYNC
    __builtin_amdgcn_global_load_async_to_lds_b128(
        (AS1 gv4i*)(A + (size_t)(bm + crow) * K + kb + ckof),
        (AS3 gv4i*)&As[buf][crow * LDT + ckof], 0, 0);
#pragma unroll
    for (int i = 0; i < 4; ++i)
      __builtin_amdgcn_global_load_async_to_lds_b128(
          (AS1 gv4i*)(Bt + (size_t)(bn + crow + 64 * i) * K + kb + ckof),
          (AS3 gv4i*)&Bs[buf][(crow + 64 * i) * LDT + ckof], 0, 0);
#else
    *reinterpret_cast<v8bf*>(&As[buf][crow * LDT + ckof]) =
        *reinterpret_cast<const v8bf*>(A + (size_t)(bm + crow) * K + kb + ckof);
#pragma unroll
    for (int i = 0; i < 4; ++i)
      *reinterpret_cast<v8bf*>(&Bs[buf][(crow + 64 * i) * LDT + ckof]) =
          *reinterpret_cast<const v8bf*>(Bt + (size_t)(bn + crow + 64 * i) * K +
                                         kb + ckof);
#endif
  };

  issue_copy(0, 0);
  int buf = 0;
  for (int kb = 0; kb < K; kb += 32, buf ^= 1) {
    wait_async0();        // own async copies for this buffer done
    __syncthreads();      // everyone's copies done & prev tile consumed
    if (kb + 32 < K) issue_copy(buf ^ 1, kb + 32);

    v16bf afr[2], bfr[4];
#pragma unroll
    for (int i = 0; i < 2; ++i) {
      const int m = wr * 32 + i * 16 + lm;
      union { v16bf v; v8bf h[2]; } u;
      u.h[0] = *reinterpret_cast<const v8bf*>(&As[buf][m * LDT + lk]);
      u.h[1] = *reinterpret_cast<const v8bf*>(&As[buf][m * LDT + lk + 16]);
      afr[i] = u.v;
    }
#pragma unroll
    for (int j = 0; j < 4; ++j) {
      const int n = wc * 64 + j * 16 + lm;
      union { v16bf v; v8bf h[2]; } u;
      u.h[0] = *reinterpret_cast<const v8bf*>(&Bs[buf][n * LDT + lk]);
      u.h[1] = *reinterpret_cast<const v8bf*>(&Bs[buf][n * LDT + lk + 16]);
      bfr[j] = u.v;
    }
#pragma unroll
    for (int i = 0; i < 2; ++i)
#pragma unroll
      for (int j = 0; j < 4; ++j)
        acc[i][j] = __builtin_amdgcn_wmma_f32_16x16x32_bf16(
            false, afr[i], false, bfr[j], (short)0, acc[i][j], false, false);
  }

  // epilogue: C/D layout — VGPR r: lanes 0-15 -> M=r, lanes 16-31 -> M=r+8
#pragma unroll
  for (int i = 0; i < 2; ++i) {
#pragma unroll
    for (int j = 0; j < 4; ++j) {
      const int col = bn + wc * 64 + j * 16 + lm;
      const float bv = (MODE >= 1) ? bias[col] : 0.0f;
      const size_t base =
          (size_t)(bm + wr * 32 + i * 16 + ((lane < 16) ? 0 : 8)) * N + col;
#pragma unroll
      for (int r = 0; r < 8; ++r) {
        float v = acc[i][j][r] + bv;
        if (MODE == 2) v = 0.5f * v * (1.0f + erff(v * 0.70710678118f));
        const size_t idx = base + (size_t)r * N;
        if (MODE == 3)
          Cf[idx] += v;
        else if (OUTBF)
          Cb[idx] = (bf16)v;
        else
          Cf[idx] = v;
      }
    }
  }
}

// ---------------------------------------------------------------------------
// Weight convert f32[K][N] -> bf16[N][K] (transpose via 32x32 LDS tile).
// Done per-layer; ~0.5 GB total traffic per pass (~22 us at HBM peak).
// ---------------------------------------------------------------------------
__global__ __launch_bounds__(256) void cvt_tr_k(const float* __restrict__ in,
                                                bf16* __restrict__ out, int K,
                                                int N) {
  __shared__ float t[32][33];
  const int k0 = blockIdx.y * 32, n0 = blockIdx.x * 32;
  const int tx = threadIdx.x & 31, ty = threadIdx.x >> 5;  // 32 x 8
#pragma unroll
  for (int i = 0; i < 32; i += 8)
    t[ty + i][tx] = in[(size_t)(k0 + ty + i) * N + n0 + tx];
  __syncthreads();
#pragma unroll
  for (int i = 0; i < 32; i += 8)
    out[(size_t)(n0 + ty + i) * K + k0 + tx] = (bf16)t[tx][ty + i];
}

// ---------------------------------------------------------------------------
// LayerNorm over DIM=768, one block (256 thr) per row; OUT = float or bf16
// ---------------------------------------------------------------------------
template <typename OUT>
__global__ __launch_bounds__(256) void layernorm_k(const float* __restrict__ X,
                                                   const float* __restrict__ g,
                                                   const float* __restrict__ b,
                                                   OUT* __restrict__ Y) {
  __shared__ float red[256];
  const int t = threadIdx.x;
  const float* x = X + (size_t)blockIdx.x * DIM;
  float v0 = x[t], v1 = x[t + 256], v2 = x[t + 512];
  red[t] = v0 + v1 + v2;
  __syncthreads();
  for (int s = 128; s > 0; s >>= 1) {
    if (t < s) red[t] += red[t + s];
    __syncthreads();
  }
  const float mu = red[0] * (1.0f / DIM);
  __syncthreads();
  const float d0 = v0 - mu, d1 = v1 - mu, d2 = v2 - mu;
  red[t] = d0 * d0 + d1 * d1 + d2 * d2;
  __syncthreads();
  for (int s = 128; s > 0; s >>= 1) {
    if (t < s) red[t] += red[t + s];
    __syncthreads();
  }
  const float rstd = rsqrtf(red[0] * (1.0f / DIM) + 1e-5f);
  OUT* y = Y + (size_t)blockIdx.x * DIM;
  y[t] = (OUT)(d0 * rstd * g[t] + b[t]);
  y[t + 256] = (OUT)(d1 * rstd * g[t + 256] + b[t + 256]);
  y[t + 512] = (OUT)(d2 * rstd * g[t + 512] + b[t + 512]);
}

// ---------------------------------------------------------------------------
// Patchify + LayerNorm(ln_p) fused -> bf16 GEMM operand [MPATCH, PD]
// row = b*196 + h*14 + w ; col = (h1*16 + w1)*3 + c
// ---------------------------------------------------------------------------
__global__ __launch_bounds__(256) void patchify_ln_k(
    const float* __restrict__ x, const float* __restrict__ g,
    const float* __restrict__ b, bf16* __restrict__ Y) {
  __shared__ float red[256];
  const int t = threadIdx.x;
  const int row = blockIdx.x;
  const int bi = row / NPATCH;
  const int pr = row % NPATCH;
  const int ph = pr / HP, pw = pr % HP;
  float v[3];
#pragma unroll
  for (int q = 0; q < 3; ++q) {
    const int col = t + q * 256;
    const int c = col % CH;
    const int pix = col / CH;
    const int w1 = pix % PATCH, h1 = pix / PATCH;
    v[q] = x[(((size_t)bi * CH + c) * IMG + (ph * PATCH + h1)) * IMG +
             (pw * PATCH + w1)];
  }
  red[t] = v[0] + v[1] + v[2];
  __syncthreads();
  for (int s = 128; s > 0; s >>= 1) {
    if (t < s) red[t] += red[t + s];
    __syncthreads();
  }
  const float mu = red[0] * (1.0f / PD);
  __syncthreads();
  const float d0 = v[0] - mu, d1 = v[1] - mu, d2 = v[2] - mu;
  red[t] = d0 * d0 + d1 * d1 + d2 * d2;
  __syncthreads();
  for (int s = 128; s > 0; s >>= 1) {
    if (t < s) red[t] += red[t + s];
    __syncthreads();
  }
  const float rstd = rsqrtf(red[0] * (1.0f / PD) + 1e-5f);
  bf16* y = Y + (size_t)row * PD;
  y[t] = (bf16)(d0 * rstd * g[t] + b[t]);
  y[t + 256] = (bf16)(d1 * rstd * g[t + 256] + b[t + 256]);
  y[t + 512] = (bf16)(d2 * rstd * g[t + 512] + b[t + 512]);
}

// ---------------------------------------------------------------------------
// z[b,0,:] = cls + pos[0]; z[b,t,:] = pe[b,t-1,:] + pos[t]
// ---------------------------------------------------------------------------
__global__ void build_z_k(const float* __restrict__ pe,
                          const float* __restrict__ cls,
                          const float* __restrict__ pos,
                          float* __restrict__ z) {
  const size_t total = (size_t)MTOK * DIM;
  for (size_t i = (size_t)blockIdx.x * blockDim.x + threadIdx.x; i < total;
       i += (size_t)gridDim.x * blockDim.x) {
    const int d = (int)(i % DIM);
    const size_t tok = i / DIM;
    const int tt = (int)(tok % NTOK);
    const int bb = (int)(tok / NTOK);
    float v = (tt == 0) ? cls[d]
                        : pe[((size_t)bb * NPATCH + (tt - 1)) * DIM + d];
    z[i] = v + pos[(size_t)tt * DIM + d];
  }
}

// ---------------------------------------------------------------------------
// Attention: one block per (query n, head h, batch b). Q/K/V f32 [MTOK,DIM],
// heads interleaved (col = h*64+d). Writes O as bf16 (next GEMM operand).
// Only ~4% of total FLOPs -> plain f32 VALU.
// ---------------------------------------------------------------------------
__global__ __launch_bounds__(128) void attention_k(const float* __restrict__ Q,
                                                   const float* __restrict__ K,
                                                   const float* __restrict__ V,
                                                   bf16* __restrict__ O) {
  __shared__ float qs[HD];
  __shared__ float sc[NTOK];
  __shared__ float red[128];
  const int n = blockIdx.x, h = blockIdx.y, b = blockIdx.z;
  const int t = threadIdx.x;
  const size_t base = (size_t)b * NTOK * DIM + (size_t)h * HD;
  if (t < HD) qs[t] = Q[base + (size_t)n * DIM + t] * 0.125f;  // 64^-0.5
  __syncthreads();
  for (int m = t; m < NTOK; m += 128) {
    const float* kp = K + base + (size_t)m * DIM;
    float s = 0.0f;
#pragma unroll
    for (int d = 0; d < HD; ++d) s += qs[d] * kp[d];
    sc[m] = s;
  }
  __syncthreads();
  float lmax = -1e30f;
  for (int m = t; m < NTOK; m += 128) lmax = fmaxf(lmax, sc[m]);
  red[t] = lmax;
  __syncthreads();
  for (int s = 64; s > 0; s >>= 1) {
    if (t < s) red[t] = fmaxf(red[t], red[t + s]);
    __syncthreads();
  }
  const float mx = red[0];
  __syncthreads();
  float lsum = 0.0f;
  for (int m = t; m < NTOK; m += 128) {
    const float e = __expf(sc[m] - mx);
    sc[m] = e;
    lsum += e;
  }
  red[t] = lsum;
  __syncthreads();
  for (int s = 64; s > 0; s >>= 1) {
    if (t < s) red[t] += red[t + s];
    __syncthreads();
  }
  const float inv = 1.0f / red[0];
  __syncthreads();
  if (t < HD) {
    float o = 0.0f;
    const float* vp = V + base + t;
    for (int m = 0; m < NTOK; ++m) o += sc[m] * vp[(size_t)m * DIM];
    O[base + (size_t)n * DIM + t] = (bf16)(o * inv);
  }
}

// ---------------------------------------------------------------------------
// Head: LN(ln_out) on CLS row, @ W_head[768,1000] + b_head, softmax -> out
// ---------------------------------------------------------------------------
__global__ __launch_bounds__(256) void head_k(
    const float* __restrict__ z, const float* __restrict__ g,
    const float* __restrict__ b, const float* __restrict__ Wh,
    const float* __restrict__ bh, float* __restrict__ out) {
  __shared__ float xr[DIM];
  __shared__ float red[256];
  __shared__ float lg[NCLS];
  const int t = threadIdx.x;
  const float* x = z + (size_t)blockIdx.x * NTOK * DIM;  // CLS token row
  float v0 = x[t], v1 = x[t + 256], v2 = x[t + 512];
  red[t] = v0 + v1 + v2;
  __syncthreads();
  for (int s = 128; s > 0; s >>= 1) {
    if (t < s) red[t] += red[t + s];
    __syncthreads();
  }
  const float mu = red[0] * (1.0f / DIM);
  __syncthreads();
  const float d0 = v0 - mu, d1 = v1 - mu, d2 = v2 - mu;
  red[t] = d0 * d0 + d1 * d1 + d2 * d2;
  __syncthreads();
  for (int s = 128; s > 0; s >>= 1) {
    if (t < s) red[t] += red[t + s];
    __syncthreads();
  }
  const float rstd = rsqrtf(red[0] * (1.0f / DIM) + 1e-5f);
  xr[t] = d0 * rstd * g[t] + b[t];
  xr[t + 256] = d1 * rstd * g[t + 256] + b[t + 256];
  xr[t + 512] = d2 * rstd * g[t + 512] + b[t + 512];
  __syncthreads();
  for (int c = t; c < NCLS; c += 256) {
    float s = bh[c];
    for (int d = 0; d < DIM; ++d) s += xr[d] * Wh[(size_t)d * NCLS + c];
    lg[c] = s;
  }
  __syncthreads();
  float lmax = -1e30f;
  for (int c = t; c < NCLS; c += 256) lmax = fmaxf(lmax, lg[c]);
  red[t] = lmax;
  __syncthreads();
  for (int s = 128; s > 0; s >>= 1) {
    if (t < s) red[t] = fmaxf(red[t], red[t + s]);
    __syncthreads();
  }
  const float mx = red[0];
  __syncthreads();
  float lsum = 0.0f;
  for (int c = t; c < NCLS; c += 256) {
    const float e = __expf(lg[c] - mx);
    lg[c] = e;
    lsum += e;
  }
  red[t] = lsum;
  __syncthreads();
  for (int s = 128; s > 0; s >>= 1) {
    if (t < s) red[t] += red[t + s];
    __syncthreads();
  }
  const float inv = 1.0f / red[0];
  __syncthreads();
  for (int c = t; c < NCLS; c += 256)
    out[(size_t)blockIdx.x * NCLS + c] = lg[c] * inv;
}

// ---------------------------------------------------------------------------
extern "C" void kernel_launch(void* const* d_in, const int* in_sizes, int n_in,
                              void* d_out, int out_size, void* d_ws,
                              size_t ws_size, hipStream_t stream) {
  (void)in_sizes; (void)n_in; (void)out_size; (void)ws_size;
  const float* x        = (const float*)d_in[0];
  const float* ln_p_g   = (const float*)d_in[1];
  const float* ln_p_b   = (const float*)d_in[2];
  const float* W_patch  = (const float*)d_in[3];
  const float* b_patch  = (const float*)d_in[4];
  const float* ln_e_g   = (const float*)d_in[5];
  const float* ln_e_b   = (const float*)d_in[6];
  const float* pos_emb  = (const float*)d_in[7];
  const float* cls_tok  = (const float*)d_in[8];
  const float* ln_a_g   = (const float*)d_in[9];
  const float* ln_a_b   = (const float*)d_in[10];
  const float* Wq       = (const float*)d_in[11];
  const float* Wk       = (const float*)d_in[12];
  const float* Wv       = (const float*)d_in[13];
  const float* Wo       = (const float*)d_in[14];
  const float* bo       = (const float*)d_in[15];
  const float* ln_f_g   = (const float*)d_in[16];
  const float* ln_f_b   = (const float*)d_in[17];
  const float* W1       = (const float*)d_in[18];
  const float* b1       = (const float*)d_in[19];
  const float* W2       = (const float*)d_in[20];
  const float* b2       = (const float*)d_in[21];
  const float* ln_out_g = (const float*)d_in[22];
  const float* ln_out_b = (const float*)d_in[23];
  const float* W_head   = (const float*)d_in[24];
  const float* b_head   = (const float*)d_in[25];

  char* ws = (char*)d_ws;
  auto alloc = [&](size_t bytes) {
    char* p = ws;
    ws += (bytes + 255) & ~(size_t)255;
    return p;
  };
  float* z   = (float*)alloc((size_t)MTOK * DIM * 4);  // residual stream
  bf16*  Xbf = (bf16*) alloc((size_t)MTOK * DIM * 2);  // LN out / patches
  float* Qf  = (float*)alloc((size_t)MTOK * DIM * 4);
  float* Kf  = (float*)alloc((size_t)MTOK * DIM * 4);
  float* Vf  = (float*)alloc((size_t)MTOK * DIM * 4);
  bf16*  Obf = (bf16*) alloc((size_t)MTOK * DIM * 2);
  bf16*  H1  = (bf16*) alloc((size_t)MTOK * HID * 2);
  bf16*  Wbf = (bf16*) alloc((size_t)DIM * HID * 2);   // transposed weight buf

  auto gg = [](int M, int N) { return dim3(N / BN, M / BM); };
  auto tg = [](int K, int N) { return dim3(N / 32, K / 32); };  // cvt_tr grid

  // --- patch embed ---
  patchify_ln_k<<<MPATCH, 256, 0, stream>>>(x, ln_p_g, ln_p_b, Xbf);
  cvt_tr_k<<<tg(PD, DIM), 256, 0, stream>>>(W_patch, Wbf, PD, DIM);
  gemm_bf16_wmma<1, false><<<gg(MPATCH, DIM), 256, 0, stream>>>(
      Xbf, Wbf, b_patch, Qf, nullptr, MPATCH, DIM, PD);
  layernorm_k<float><<<MPATCH, 256, 0, stream>>>(Qf, ln_e_g, ln_e_b, Kf);
  build_z_k<<<2048, 256, 0, stream>>>(Kf, cls_tok, pos_emb, z);

  // --- transformer layers ---
  for (int i = 0; i < DEPTH; ++i) {
    const size_t wdd = (size_t)i * DIM * DIM;
    const size_t wdh = (size_t)i * DIM * HID;
    layernorm_k<bf16><<<MTOK, 256, 0, stream>>>(z, ln_a_g + i * DIM,
                                                ln_a_b + i * DIM, Xbf);
    cvt_tr_k<<<tg(DIM, DIM), 256, 0, stream>>>(Wq + wdd, Wbf, DIM, DIM);
    gemm_bf16_wmma<0, false><<<gg(MTOK, DIM), 256, 0, stream>>>(
        Xbf, Wbf, nullptr, Qf, nullptr, MTOK, DIM, DIM);
    cvt_tr_k<<<tg(DIM, DIM), 256, 0, stream>>>(Wk + wdd, Wbf, DIM, DIM);
    gemm_bf16_wmma<0, false><<<gg(MTOK, DIM), 256, 0, stream>>>(
        Xbf, Wbf, nullptr, Kf, nullptr, MTOK, DIM, DIM);
    cvt_tr_k<<<tg(DIM, DIM), 256, 0, stream>>>(Wv + wdd, Wbf, DIM, DIM);
    gemm_bf16_wmma<0, false><<<gg(MTOK, DIM), 256, 0, stream>>>(
        Xbf, Wbf, nullptr, Vf, nullptr, MTOK, DIM, DIM);
    attention_k<<<dim3(NTOK, HEADS, BATCH), 128, 0, stream>>>(Qf, Kf, Vf, Obf);
    cvt_tr_k<<<tg(DIM, DIM), 256, 0, stream>>>(Wo + wdd, Wbf, DIM, DIM);
    gemm_bf16_wmma<3, false><<<gg(MTOK, DIM), 256, 0, stream>>>(
        Obf, Wbf, bo + i * DIM, z, nullptr, MTOK, DIM, DIM);  // z += o@Wo+bo
    layernorm_k<bf16><<<MTOK, 256, 0, stream>>>(z, ln_f_g + i * DIM,
                                                ln_f_b + i * DIM, Xbf);
    cvt_tr_k<<<tg(DIM, HID), 256, 0, stream>>>(W1 + wdh, Wbf, DIM, HID);
    gemm_bf16_wmma<2, true><<<gg(MTOK, HID), 256, 0, stream>>>(
        Xbf, Wbf, b1 + i * HID, nullptr, H1, MTOK, HID, DIM);  // gelu -> bf16
    cvt_tr_k<<<tg(HID, DIM), 256, 0, stream>>>(W2 + wdh, Wbf, HID, DIM);
    gemm_bf16_wmma<3, false><<<gg(MTOK, DIM), 256, 0, stream>>>(
        H1, Wbf, b2 + i * DIM, z, nullptr, MTOK, DIM, HID);  // z += h@W2+b2
  }

  // --- classification head + softmax ---
  head_k<<<BATCH, 256, 0, stream>>>(z, ln_out_g, ln_out_b, W_head, b_head,
                                    (float*)d_out);
}
